// GCNEncoder_38276748542405
// MI455X (gfx1250) — matrix operand order repaired and checked by reading, not compiled, not run on hardware
//
#include <hip/hip_runtime.h>

// ---------------------------------------------------------------------------
// DGCNN-style encoder for MI455X (gfx1250, wave32, WMMA).
// All GEMMs (1x1 convs and KNN distance matrices) run through
// v_wmma_f32_16x16x32_bf16 with fp32 accumulation.
// Each wave computes a 16x64 tile: per k-step, 1 A-fragment + 4 B-fragments
// are loaded up front (A: 2x b128; B: strided 16-bit loads), then 4 WMMAs issue.  The B
// stream is prefetched 2 k-steps ahead (unconditional, non-faulting).
//
// Workspace requirement: ~497 MB (see bump allocator in kernel_launch).
// ---------------------------------------------------------------------------

typedef __attribute__((ext_vector_type(16))) __bf16        v16bf;
typedef __attribute__((ext_vector_type(8)))  float         v8f;
typedef __attribute__((ext_vector_type(4))) unsigned int   u32x4;

union FragU { u32x4 u[2]; v16bf v; };

#define BB      8
#define NN      2048
#define CC0     64
#define KNN     10
#define LL1     (BB * NN * KNN)   // 163840
#define BN      (BB * NN)         // 16384
#define NSP     2048

static __device__ __forceinline__ unsigned short f2bf(float f) {
  unsigned int u = __float_as_uint(f);
  u += 0x7FFFu + ((u >> 16) & 1u);      // round-to-nearest-even
  return (unsigned short)(u >> 16);
}
static __device__ __forceinline__ float bf2f(unsigned short s) {
  return __uint_as_float(((unsigned int)s) << 16);
}

// ---------------------------------------------------------------------------
// Pack fp32 weights into zero-padded bf16 (Opad x Ipad), Opad mult of 64,
// Ipad mult of 32, so WMMA tiles never read out of bounds.
// ---------------------------------------------------------------------------
__global__ void convert_w_kernel(const float* __restrict__ w, int O, int I,
                                 unsigned short* __restrict__ Wb, int Opad,
                                 int Ipad) {
  int t = blockIdx.x * blockDim.x + threadIdx.x;
  if (t >= Opad * Ipad) return;
  int o = t / Ipad, i = t % Ipad;
  float v = (o < O && i < I) ? w[o * I + i] : 0.0f;
  Wb[t] = f2bf(v);
}

__global__ void zero_u16_kernel(unsigned short* __restrict__ p, long n) {
  long t = (long)blockIdx.x * blockDim.x + threadIdx.x;
  if (t < n) p[t] = 0;
}

// ---------------------------------------------------------------------------
// Per-batch point packing: channel-major fp32 -> point-major bf16 (N x Cpad)
// for A-fragments, transposed bf16 (Cpad x N) for B-fragments, plus |x|^2.
// ---------------------------------------------------------------------------
__global__ void pack_points_kernel(const float* __restrict__ src, int C,
                                   long cstride, long base,
                                   unsigned short* __restrict__ Xp,
                                   unsigned short* __restrict__ XpT, int Cpad,
                                   float* __restrict__ sq, int N) {
  int n = blockIdx.x * blockDim.x + threadIdx.x;
  if (n >= N) return;
  float s = 0.0f;
  for (int c = 0; c < Cpad; ++c) {
    float v = 0.0f;
    if (c < C) {
      v = src[(long)c * cstride + base + n];
      s += v * v;
    }
    unsigned short bv = f2bf(v);
    Xp[(long)n * Cpad + c] = bv;
    XpT[(long)c * N + n]   = bv;
  }
  sq[n] = s;
}

// ---------------------------------------------------------------------------
// Shared WMMA 16x64-per-wave loop: A (row-major, lda) x B (row-major, ldb).
// A frag per ISA 7.12.2 (lane-half h: K in {h*8.., 16+h*8..}),
// B frag: lane holds column n=lane&15, K in {h*8.., 16+h*8..} (strided).
// ---------------------------------------------------------------------------
static __device__ __forceinline__ void wmma_16x64_loop(
    const unsigned short* __restrict__ A, int lda,
    const unsigned short* __restrict__ B, long ldb, int Kdim, int mBase,
    int nBase, int lane, v8f acc[4]) {
  const unsigned short* ap =
      A + (long)(mBase + (lane & 15)) * lda + (lane >> 4) * 8;
  const unsigned short* bp = B + (long)((lane >> 4) * 8) * ldb + nBase + (lane & 15);
  const long bstep = 32 * ldb;
  for (int k0 = 0; k0 < Kdim; k0 += 32, ap += 32, bp += bstep) {
    FragU ua;
    ua.u[0] = *reinterpret_cast<const u32x4*>(ap);        // K = h*8..h*8+7
    ua.u[1] = *reinterpret_cast<const u32x4*>(ap + 16);   // K = 16+h*8..
    // Unconditional B-stream prefetch 2 k-steps ahead (global_prefetch_b8,
    // non-faulting; overshoot stays inside the workspace).
    __builtin_prefetch(bp + 2 * bstep, 0, 1);
    FragU ub[4];
#pragma unroll
    for (int t = 0; t < 4; ++t) {  // issue all B loads before WMMAs
      for (int j = 0; j < 4; ++j) {
        ub[t].u[0][j] = (unsigned)bp[t * 16 + (2 * j) * ldb] |
                        ((unsigned)bp[t * 16 + (2 * j + 1) * ldb] << 16);
        ub[t].u[1][j] = (unsigned)bp[t * 16 + (16 + 2 * j) * ldb] |
                        ((unsigned)bp[t * 16 + (17 + 2 * j) * ldb] << 16);
      }
    }
#pragma unroll
    for (int t = 0; t < 4; ++t)
      acc[t] = __builtin_amdgcn_wmma_f32_16x16x32_bf16(
          false, ua.v, false, ub[t].v, (short)0, acc[t], false, false);
  }
}

// ---------------------------------------------------------------------------
// D[i,j] = |xi|^2 + |xj|^2 - 2 * <xi,xj>  via WMMA bf16.
// A from Xp (N x Cpad), B from XpT (Cpad x N) -> b128 loads for A, strided 16-bit loads for B.
// 8 waves/block (4M x 2N): block tile 64x128, wave tile 16x64.
// ---------------------------------------------------------------------------
__global__ void dist_gemm_kernel(const unsigned short* __restrict__ Xp,
                                 const unsigned short* __restrict__ XpT,
                                 const float* __restrict__ sq,
                                 float* __restrict__ D, int N, int Cpad) {
  int lane = threadIdx.x & 31, wid = threadIdx.x >> 5;
  int iBase = blockIdx.y * 64 + (wid & 3) * 16;
  int jBase = blockIdx.x * 128 + (wid >> 2) * 64;
  v8f acc[4] = {};
  wmma_16x64_loop(Xp, Cpad, XpT, (long)N, Cpad, iBase, jBase, lane, acc);
  int rbase = iBase + ((lane >> 4) << 3);
#pragma unroll
  for (int t = 0; t < 4; ++t) {
    int col = jBase + t * 16 + (lane & 15);
    float sqc = sq[col];
#pragma unroll
    for (int r = 0; r < 8; ++r) {
      int row = rbase + r;
      D[(long)row * N + col] = sq[row] + sqc - 2.0f * acc[t][r];
    }
  }
}

// ---------------------------------------------------------------------------
// Per-row top-10 smallest distances (stable on ties -> matches lax.top_k).
// ---------------------------------------------------------------------------
__global__ void topk_kernel(const float* __restrict__ D, int N,
                            int* __restrict__ idxOut) {
  int i = blockIdx.x * blockDim.x + threadIdx.x;
  if (i >= N) return;
  float bd[KNN];
  int   bi[KNN];
#pragma unroll
  for (int k = 0; k < KNN; ++k) { bd[k] = 3.4e38f; bi[k] = 0; }
  const float* row = D + (long)i * N;
  for (int j = 0; j < N; ++j) {
    float d = row[j];
    if (d < bd[KNN - 1]) {
      int p = KNN - 1;
      while (p > 0 && d < bd[p - 1]) {
        bd[p] = bd[p - 1];
        bi[p] = bi[p - 1];
        --p;
      }
      bd[p] = d;
      bi[p] = j;
    }
  }
#pragma unroll
  for (int k = 0; k < KNN; ++k) idxOut[i * KNN + k] = bi[k];
}

// ---------------------------------------------------------------------------
// Edge features: rows [0,C) = nei - cen ; rows [C,2C) = cen. bf16 output
// (2C x L) row-major, l = (b*N + n)*KNN + k.
// ---------------------------------------------------------------------------
__global__ void edge_feat_kernel(const float* __restrict__ src, long cstride,
                                 long bstride, const int* __restrict__ knnidx,
                                 unsigned short* __restrict__ F, int C, long L,
                                 int N, int K) {
  long t = (long)blockIdx.x * blockDim.x + threadIdx.x;
  if (t >= (long)C * L) return;
  int  c  = (int)(t / L);
  long l  = t % L;
  long bn = l / K;
  int  b  = (int)(bn / N);
  int  n  = (int)(bn % N);
  int  j  = knnidx[l];
  float cen = src[(long)c * cstride + (long)b * bstride + n];
  float nei = src[(long)c * cstride + (long)b * bstride + j];
  F[(long)c * L + l]       = f2bf(nei - cen);
  F[(long)(C + c) * L + l] = f2bf(cen);
}

// ---------------------------------------------------------------------------
// GEMM + bias + BN + tanh-GELU.  W (Opad x Ipad bf16), X (>=Ipad x L bf16).
// 8 waves/block (4M x 2N): block tile 64x128, wave tile 16x64.
// mode 0: bf16 row-major (O x L);  mode 1: fp32 [(b*O+o)*Nsp + n] layout.
// ---------------------------------------------------------------------------
__global__ void gemm_bn_gelu_kernel(const unsigned short* __restrict__ Wb,
                                    int Ipad, const unsigned short* __restrict__ X,
                                    int L, int O, const float* __restrict__ bias,
                                    const float* __restrict__ gamma,
                                    const float* __restrict__ beta,
                                    unsigned short* __restrict__ Ybf,
                                    float* __restrict__ Yout, int mode,
                                    int Nsp) {
  int lane = threadIdx.x & 31, wid = threadIdx.x >> 5;
  int mBase = blockIdx.y * 64 + (wid & 3) * 16;
  int nBase = blockIdx.x * 128 + (wid >> 2) * 64;
  v8f acc[4] = {};
  wmma_16x64_loop(Wb, Ipad, X, (long)L, Ipad, mBase, nBase, lane, acc);
  int rbase = mBase + ((lane >> 4) << 3);
  const float inv = rsqrtf(1.0f + 1e-5f);
#pragma unroll
  for (int t = 0; t < 4; ++t) {
    int col = nBase + t * 16 + (lane & 15);
#pragma unroll
    for (int r = 0; r < 8; ++r) {
      int row = rbase + r;
      if (row < O) {
        float y = (acc[t][r] + bias[row]) * (gamma[row] * inv) + beta[row];
        float g =
            0.5f * y *
            (1.0f + tanhf(0.7978845608028654f * (y + 0.044715f * y * y * y)));
        if (mode == 0) {
          Ybf[(long)row * L + col] = f2bf(g);
        } else {
          int bb = col / Nsp, n = col % Nsp;
          Yout[((long)bb * O + row) * Nsp + n] = g;
        }
      }
    }
  }
}

// ---------------------------------------------------------------------------
// Max over K neighbors: (O x L) bf16 -> (O x BN).  Writes fp32 (optional)
// and bf16 (into concat buffer at caller-chosen row offset).
// ---------------------------------------------------------------------------
__global__ void max_over_k_kernel(const unsigned short* __restrict__ F, int O,
                                  long L, int K, float* __restrict__ outF,
                                  unsigned short* __restrict__ outB, int bn) {
  long t = (long)blockIdx.x * blockDim.x + threadIdx.x;
  if (t >= (long)O * bn) return;
  int o = (int)(t / bn), q = (int)(t % bn);
  const unsigned short* p = F + (long)o * L + (long)q * K;
  float m = -3.4e38f;
#pragma unroll
  for (int k = 0; k < KNN; ++k) m = fmaxf(m, bf2f(p[k]));
  if (outF) outF[(long)o * bn + q] = m;
  outB[(long)o * bn + q] = f2bf(m);
}

// ---------------------------------------------------------------------------
extern "C" void kernel_launch(void* const* d_in, const int* in_sizes, int n_in,
                              void* d_out, int out_size, void* d_ws,
                              size_t ws_size, hipStream_t stream) {
  const float* x   = (const float*)d_in[0];
  const float* w11 = (const float*)d_in[1],  *b11 = (const float*)d_in[2],
             * g11 = (const float*)d_in[3],  *t11 = (const float*)d_in[4];
  const float* w12 = (const float*)d_in[5],  *b12 = (const float*)d_in[6],
             * g12 = (const float*)d_in[7],  *t12 = (const float*)d_in[8];
  const float* w21 = (const float*)d_in[9],  *b21 = (const float*)d_in[10],
             * g21 = (const float*)d_in[11], *t21 = (const float*)d_in[12];
  const float* w22 = (const float*)d_in[13], *b22 = (const float*)d_in[14],
             * g22 = (const float*)d_in[15], *t22 = (const float*)d_in[16];
  const float* w31 = (const float*)d_in[17], *b31 = (const float*)d_in[18],
             * g31 = (const float*)d_in[19], *t31 = (const float*)d_in[20];
  const float* w32 = (const float*)d_in[21], *b32 = (const float*)d_in[22],
             * g32 = (const float*)d_in[23], *t32 = (const float*)d_in[24];
  float* out = (float*)d_out;

  // ---- workspace bump allocator (total ~497 MB) ----
  char* ws = (char*)d_ws;
  size_t off = 0;
  auto alloc = [&](size_t bytes) -> char* {
    char* p = ws + off;
    off = (off + bytes + 255) & ~(size_t)255;
    return p;
  };
  float*          D    = (float*)alloc((size_t)NN * NN * 4);          // 16 MB
  unsigned short* Xp   = (unsigned short*)alloc((size_t)NN * 192 * 2);
  unsigned short* XpT  = (unsigned short*)alloc((size_t)192 * NN * 2);
  float*          sq   = (float*)alloc((size_t)NN * 4);
  int*            idx1 = (int*)alloc((size_t)LL1 * 4);
  int*            idx2 = (int*)alloc((size_t)LL1 * 4);
  unsigned short* Wbuf = (unsigned short*)alloc((size_t)640 * 704 * 2);
  float*          x1f  = (float*)alloc((size_t)181 * BN * 4);
  unsigned short* Xcat = (unsigned short*)alloc((size_t)704 * BN * 2);
  // Reused big regions: RegA {F0,F4}, RegB {F1,F5}, RegC {F2,F3,F6}
  unsigned short* RegA = (unsigned short*)alloc((size_t)448 * LL1 * 2); // 147 MB
  unsigned short* RegB = (unsigned short*)alloc((size_t)512 * LL1 * 2); // 168 MB
  unsigned short* RegC = (unsigned short*)alloc((size_t)384 * LL1 * 2); // 126 MB
  unsigned short* F0 = RegA, *F4 = RegA;
  unsigned short* F1 = RegB, *F5 = RegB;
  unsigned short* F2 = RegC, *F3 = RegC, *F6 = RegC;

  const int T = 256;
  auto blocks = [](long n) { return (unsigned)((n + 255) / 256); };

  // ================= layer 1: KNN on x (C=64) =================
  for (int b = 0; b < BB; ++b) {
    pack_points_kernel<<<blocks(NN), T, 0, stream>>>(
        x, CC0, (long)NN, (long)b * CC0 * NN, Xp, XpT, 64, sq, NN);
    dist_gemm_kernel<<<dim3(NN / 128, NN / 64), T, 0, stream>>>(Xp, XpT, sq, D,
                                                                NN, 64);
    topk_kernel<<<blocks(NN), T, 0, stream>>>(D, NN, idx1 + (size_t)b * NN * KNN);
  }
  edge_feat_kernel<<<blocks((long)CC0 * LL1), T, 0, stream>>>(
      x, (long)NN, (long)CC0 * NN, idx1, F0, CC0, LL1, NN, KNN);

  // conv11: (152 x 128) -> F1 (bf16, 160 rows padded)
  convert_w_kernel<<<blocks(192 * 128), T, 0, stream>>>(w11, 152, 128, Wbuf, 192, 128);
  gemm_bn_gelu_kernel<<<dim3(LL1 / 128, 192 / 64), T, 0, stream>>>(
      Wbuf, 128, F0, LL1, 152, b11, g11, t11, F1, nullptr, 0, NSP);
  zero_u16_kernel<<<blocks((long)8 * LL1), T, 0, stream>>>(F1 + (size_t)152 * LL1, (long)8 * LL1);

  // conv12: (181 x 152 pad 160) -> F2 (bf16, 181 rows)
  convert_w_kernel<<<blocks(192 * 160), T, 0, stream>>>(w12, 181, 152, Wbuf, 192, 160);
  gemm_bn_gelu_kernel<<<dim3(LL1 / 128, 192 / 64), T, 0, stream>>>(
      Wbuf, 160, F1, LL1, 181, b12, g12, t12, F2, nullptr, 0, NSP);

  // x1 = max_k -> fp32 (181 x BN) + bf16 into Xcat rows [0,181)
  max_over_k_kernel<<<blocks((long)181 * BN), T, 0, stream>>>(
      F2, 181, (long)LL1, KNN, x1f, Xcat, BN);

  // ================= layer 2: KNN on x1 (C=181, pad 192) =================
  for (int b = 0; b < BB; ++b) {
    pack_points_kernel<<<blocks(NN), T, 0, stream>>>(
        x1f, 181, (long)BN, (long)b * NN, Xp, XpT, 192, sq, NN);
    dist_gemm_kernel<<<dim3(NN / 128, NN / 64), T, 0, stream>>>(Xp, XpT, sq, D,
                                                                NN, 192);
    topk_kernel<<<blocks(NN), T, 0, stream>>>(D, NN, idx2 + (size_t)b * NN * KNN);
  }
  edge_feat_kernel<<<blocks((long)181 * LL1), T, 0, stream>>>(
      x1f, (long)BN, (long)NN, idx2, F3, 181, LL1, NN, KNN);
  zero_u16_kernel<<<blocks((long)22 * LL1), T, 0, stream>>>(F3 + (size_t)362 * LL1, (long)22 * LL1);

  // conv21: (430 x 362 pad 384) -> F4 (bf16, 448 rows padded)
  convert_w_kernel<<<blocks(448 * 384), T, 0, stream>>>(w21, 430, 362, Wbuf, 448, 384);
  gemm_bn_gelu_kernel<<<dim3(LL1 / 128, 448 / 64), T, 0, stream>>>(
      Wbuf, 384, F3, LL1, 430, b21, g21, t21, F4, nullptr, 0, NSP);
  zero_u16_kernel<<<blocks((long)18 * LL1), T, 0, stream>>>(F4 + (size_t)430 * LL1, (long)18 * LL1);

  // conv22: (512 x 430 pad 448) -> F5 (bf16, 512 rows)
  convert_w_kernel<<<blocks(512 * 448), T, 0, stream>>>(w22, 512, 430, Wbuf, 512, 448);
  gemm_bn_gelu_kernel<<<dim3(LL1 / 128, 512 / 64), T, 0, stream>>>(
      Wbuf, 448, F4, LL1, 512, b22, g22, t22, F5, nullptr, 0, NSP);

  // x2 = max_k -> bf16 into Xcat rows [181,693)
  max_over_k_kernel<<<blocks((long)512 * BN), T, 0, stream>>>(
      F5, 512, (long)LL1, KNN, nullptr, Xcat + (size_t)181 * BN, BN);
  zero_u16_kernel<<<blocks((long)11 * BN), T, 0, stream>>>(Xcat + (size_t)693 * BN, (long)11 * BN);

  // conv31: (595 x 693 pad 704) on Xcat -> F6 (bf16, 608 rows padded)
  convert_w_kernel<<<blocks(640 * 704), T, 0, stream>>>(w31, 595, 693, Wbuf, 640, 704);
  gemm_bn_gelu_kernel<<<dim3(BN / 128, 640 / 64), T, 0, stream>>>(
      Wbuf, 704, Xcat, BN, 595, b31, g31, t31, F6, nullptr, 0, NSP);
  zero_u16_kernel<<<blocks((long)13 * BN), T, 0, stream>>>(F6 + (size_t)595 * BN, (long)13 * BN);

  // conv32: (512 x 595 pad 608) -> fp32 output, batched (b,512,2048) layout
  convert_w_kernel<<<blocks(512 * 608), T, 0, stream>>>(w32, 512, 595, Wbuf, 512, 608);
  gemm_bn_gelu_kernel<<<dim3(BN / 128, 512 / 64), T, 0, stream>>>(
      Wbuf, 608, F6, BN, 512, b32, g32, t32, nullptr, out, 1, NSP);

  (void)in_sizes; (void)n_in; (void)out_size; (void)ws_size;
}
